// SelfAttentionEdge2Node_44444321579152
// MI455X (gfx1250) — matrix operand order repaired and checked by reading, not compiled, not run on hardware
//
#include <hip/hip_runtime.h>

typedef __attribute__((ext_vector_type(16))) _Float16 v16h;
typedef __attribute__((ext_vector_type(8)))  _Float16 v8h;
typedef __attribute__((ext_vector_type(4)))  _Float16 v4h;
typedef __attribute__((ext_vector_type(8)))  float    v8f;
typedef __attribute__((ext_vector_type(4)))  float    v4f;

constexpr int kBatch = 4;
constexpr int kN     = 256;   // sequence length == hidden == head dim
constexpr int kD     = 256;
constexpr int kTM    = 128;   // edge rows per tile
constexpr int kITER  = 8;     // tiles processed per block (persistent ctx regs)
constexpr int kPB    = kN * kN / (kTM * kITER);   // 64 partial blocks per batch
#define LN_EPS 1e-5f

static __device__ __forceinline__ v8f zero8() {
  v8f z;
  #pragma unroll
  for (int i = 0; i < 8; ++i) z[i] = 0.0f;
  return z;
}

static __device__ __forceinline__ v8f wmma16(v16h a, v16h b, v8f c) {
  return __builtin_amdgcn_wmma_f32_16x16x32_f16(
      false, a, false, b, (short)0, c, false, false);
}

// A fragment (16x32 f16), row-major src, row stride ldk.
// lanes0-15 vgpr0-3 K=0..7, vgpr4-7 K=16..23; lanes16-31 shifted +8.
static __device__ __forceinline__ v16h load_fragA(const _Float16* src, int row,
                                                  int ldk, int k0) {
  const int half = (threadIdx.x & 31) >> 4;
  const _Float16* p = src + row * ldk + k0 + half * 8;
  v8h lo = *(const v8h*)p;
  v8h hi = *(const v8h*)(p + 16);
  return __builtin_shufflevector(lo, hi, 0,1,2,3,4,5,6,7,8,9,10,11,12,13,14,15);
}

// B fragment (32x16 f16) from Brow[col][k] layout, row stride ldk.
// lanes0-15 vgpr0-3 K=0..7, vgpr4-7 K=16..23; lanes16-31 shifted +8.
static __device__ __forceinline__ v16h load_fragB(const _Float16* src, int col,
                                                  int ldk, int k0) {
  const int half = (threadIdx.x & 31) >> 4;
  const _Float16* p = src + col * ldk + k0 + half * 8;
  v8h lo = *(const v8h*)p;
  v8h hi = *(const v8h*)(p + 16);
  return __builtin_shufflevector(lo, hi, 0,1,2,3,4,5,6,7,8,9,10,11,12,13,14,15);
}

// -------------------------------------------------------------------------
// Prep kernels
// -------------------------------------------------------------------------
__global__ void k_zero(float* p, int n) {
  int i = blockIdx.x * 256 + threadIdx.x;
  if (i < n) p[i] = 0.0f;
}

// k = hidden_states @ Wk + bk  -> f16 [B*N][D]
__global__ __launch_bounds__(256) void k_prep_k(const float* __restrict__ hs,
                                                const float* __restrict__ Wk,
                                                const float* __restrict__ bk,
                                                _Float16* __restrict__ Kf) {
  __shared__ float sH[kD];
  const int row = blockIdx.x;        // b*N + n
  const int d = threadIdx.x;
  sH[d] = hs[(size_t)row * kD + d];
  __syncthreads();
  float acc = bk[d];
  for (int i = 0; i < kD; ++i) acc = fmaf(sH[i], Wk[i * kD + d], acc);
  Kf[(size_t)row * kD + d] = (_Float16)acc;
}

// WT[o][i] = (f16) W[i][o]
__global__ __launch_bounds__(256) void k_transpose_f16(const float* __restrict__ W,
                                                       _Float16* __restrict__ WT) {
  const int i = blockIdx.x;
  const int o = threadIdx.x;
  WT[o * kD + i] = (_Float16)W[i * kD + o];
}

// -------------------------------------------------------------------------
// Fused main kernel (512 threads = 16 waves, persistent ctx accumulators)
// -------------------------------------------------------------------------
__global__ __launch_bounds__(512, 1) void k_edge_attn(
    const float* __restrict__ SM,        // [B, N*N, D] f32
    const float* __restrict__ mask,      // [B, N]
    const _Float16* __restrict__ Kf,     // [B, N, D] f16
    const _Float16* __restrict__ WqT,    // [out][in] f16
    const _Float16* __restrict__ WvT,    // [out][in] f16
    const float* __restrict__ bq,
    const float* __restrict__ bv,
    float* __restrict__ dst,             // atomic: ctx [B,N,D]; else partials
    int useAtomic)
{
  __shared__ _Float16 sA[kTM * kD];   // SM tile f16; reused as P^T [n][m]
  __shared__ _Float16 sQ[kTM * kD];   // q f16 [m][d]
  __shared__ _Float16 sV[kD * kTM];   // v f16 transposed [d][m]
  __shared__ float    sMask[kTM];
  __shared__ float    sRow[2 * kTM];  // partial row sums (two col halves)

  const int b    = blockIdx.y;
  const int tid  = threadIdx.x;
  const int wave = tid >> 5;          // 0..15
  const int lane = tid & 31;
  const int lc   = lane & 15;
  const int lh   = lane >> 4;
  const int rb   = wave >> 1;         // row-block 0..7 for q/v/scores
  const int ch   = wave & 1;          // column half 0/1 (8 blocks each)

  // persistent context accumulators: wave owns n-block == wave, all 16 d-blocks
  v8f ctxAcc[16];
  #pragma unroll
  for (int t = 0; t < 16; ++t) ctxAcc[t] = zero8();

  for (int it = 0; it < kITER; ++it) {
    const int m0 = (blockIdx.x * kITER + it) * kTM;

    // ---- stage SM tile as f16 (non-temporal b128 loads), mask constants ----
    {
      const v4f* smp = (const v4f*)(SM + ((size_t)b * (kN * kN) + m0) * kD);
      for (int i = tid; i < kTM * kD / 4; i += 512) {
        v4f x = __builtin_nontemporal_load(smp + i);
        v4h h;
        h[0] = (_Float16)x[0]; h[1] = (_Float16)x[1];
        h[2] = (_Float16)x[2]; h[3] = (_Float16)x[3];
        *(v4h*)&sA[i * 4] = h;
      }
      if (tid < kTM) {
        int mg = m0 + tid;
        sMask[tid] = mask[b * kN + (mg >> 8)] * mask[b * kN + (mg & (kN - 1))];
      }
    }
    __syncthreads();

    // ---- q = SM @ Wq + bq -> sQ ;  wave covers rows rb, cols ch*8..+8 ----
    {
      v8f acc[8];
      #pragma unroll
      for (int t = 0; t < 8; ++t) acc[t] = zero8();
      for (int ks = 0; ks < kD / 32; ++ks) {
        v16h a = load_fragA(sA, rb * 16 + lc, kD, ks * 32);
        #pragma unroll
        for (int j = 0; j < 8; ++j) {
          const int cb = ch * 8 + j;
          acc[j] = wmma16(a, load_fragB(WqT, cb * 16 + lc, kD, ks * 32), acc[j]);
        }
      }
      #pragma unroll
      for (int j = 0; j < 8; ++j) {
        const int d = (ch * 8 + j) * 16 + lc;
        const float bias = bq[d];
        #pragma unroll
        for (int r = 0; r < 8; ++r) {
          const int m = rb * 16 + r + 8 * lh;
          sQ[m * kD + d] = (_Float16)(acc[j][r] + bias);
        }
      }
    }

    // ---- v = SM @ Wv + bv -> sV (transposed [d][m]) ----
    {
      v8f acc[8];
      #pragma unroll
      for (int t = 0; t < 8; ++t) acc[t] = zero8();
      for (int ks = 0; ks < kD / 32; ++ks) {
        v16h a = load_fragA(sA, rb * 16 + lc, kD, ks * 32);
        #pragma unroll
        for (int j = 0; j < 8; ++j) {
          const int cb = ch * 8 + j;
          acc[j] = wmma16(a, load_fragB(WvT, cb * 16 + lc, kD, ks * 32), acc[j]);
        }
      }
      #pragma unroll
      for (int j = 0; j < 8; ++j) {
        const int d = (ch * 8 + j) * 16 + lc;
        const float bias = bv[d];
        #pragma unroll
        for (int r = 0; r < 8; ++r) {
          const int m = rb * 16 + r + 8 * lh;
          sV[d * kTM + m] = (_Float16)(acc[j][r] + bias);
        }
      }
    }
    __syncthreads();   // sA free; sQ/sV visible

    // ---- scores = q @ k^T / 16 + m2d; relu; row-normalize -> P^T in sA ----
    {
      v8f acc[8];
      #pragma unroll
      for (int t = 0; t < 8; ++t) acc[t] = zero8();
      const _Float16* kb = Kf + (size_t)b * kN * kD;
      for (int ks = 0; ks < kD / 32; ++ks) {
        v16h a = load_fragA(sQ, rb * 16 + lc, kD, ks * 32);
        #pragma unroll
        for (int j = 0; j < 8; ++j) {
          const int nb = ch * 8 + j;
          acc[j] = wmma16(a, load_fragB(kb, nb * 16 + lc, kD, ks * 32), acc[j]);
        }
      }
      // relu + partial row sums for this column half
      #pragma unroll
      for (int r = 0; r < 8; ++r) {
        const int m = rb * 16 + r + 8 * lh;
        const float mv = sMask[m];
        float rs = 0.0f;
        #pragma unroll
        for (int j = 0; j < 8; ++j) {
          float x = acc[j][r] * 0.0625f + mv;   // 1/sqrt(256)
          x = x > 0.0f ? x : 0.0f;
          acc[j][r] = x;
          rs += x;
        }
        rs += __shfl_xor(rs, 1, 32);
        rs += __shfl_xor(rs, 2, 32);
        rs += __shfl_xor(rs, 4, 32);
        rs += __shfl_xor(rs, 8, 32);
        if (lc == 0) sRow[ch * kTM + m] = rs;
      }
      __syncthreads();   // both column-half partial sums visible
      #pragma unroll
      for (int r = 0; r < 8; ++r) {
        const int m = rb * 16 + r + 8 * lh;
        const float tot = sRow[m] + sRow[kTM + m];
        const float inv = 1.0f / (tot + (float)kN * 1e-12f);
        #pragma unroll
        for (int j = 0; j < 8; ++j) {
          const int n = (ch * 8 + j) * 16 + lc;
          sA[n * kTM + m] = (_Float16)(acc[j][r] * inv);   // P^T [n][m]
        }
      }
    }
    __syncthreads();   // sP (=sA) visible

    // ---- context: ctxAcc[n-block=wave][*] += P^T @ V over this tile ----
    // Manually double-buffered so each WMMA waits only on fragments issued
    // one step earlier (s_wait_dscnt > 0 instead of 0).
    {
      v16h a    = load_fragA(sA, wave * 16 + lc, kTM, 0);
      v16h bcur = load_fragB(sV, lc, kTM, 0);
      #pragma unroll
      for (int ks = 0; ks < kTM / 32; ++ks) {
        v16h anext = a;
        if (ks + 1 < kTM / 32)
          anext = load_fragA(sA, wave * 16 + lc, kTM, (ks + 1) * 32);
        #pragma unroll
        for (int db = 0; db < 16; ++db) {
          v16h bnext = bcur;
          if (db < 15)
            bnext = load_fragB(sV, (db + 1) * 16 + lc, kTM, ks * 32);
          else if (ks + 1 < kTM / 32)
            bnext = load_fragB(sV, lc, kTM, (ks + 1) * 32);
          ctxAcc[db] = wmma16(a, bcur, ctxAcc[db]);
          bcur = bnext;
        }
        a = anext;
      }
    }
    __syncthreads();   // done reading sA/sV before next iteration's staging
  }

  // ---- emit this block's context contribution ----
  if (useAtomic) {
    float* cb_ = dst + (size_t)b * kN * kD;
    #pragma unroll
    for (int db = 0; db < 16; ++db) {
      const int d = db * 16 + lc;
      #pragma unroll
      for (int r = 0; r < 8; ++r) {
        const int n = wave * 16 + r + 8 * lh;
        atomicAdd(&cb_[n * kD + d], ctxAcc[db][r]);
      }
    }
  } else {
    float* pb = dst + ((size_t)b * kPB + blockIdx.x) * kN * kD;
    #pragma unroll
    for (int db = 0; db < 16; ++db) {
      const int d = db * 16 + lc;
      #pragma unroll
      for (int r = 0; r < 8; ++r) {
        const int n = wave * 16 + r + 8 * lh;
        pb[n * kD + d] = ctxAcc[db][r];
      }
    }
  }
}

// -------------------------------------------------------------------------
// Finalize: P-way partial reduction + out-proj + residual + LayerNorm
// (atomic path uses P==1 with src == ctx)
// -------------------------------------------------------------------------
__global__ __launch_bounds__(256) void k_final(const float* __restrict__ src,
                                               int P, size_t batchStride,
                                               const float* __restrict__ Wo,
                                               const float* __restrict__ bo,
                                               const float* __restrict__ hs,
                                               const float* __restrict__ gamma,
                                               const float* __restrict__ beta,
                                               float* __restrict__ out) {
  __shared__ float sC[kD];
  __shared__ float red[256];
  const int row = blockIdx.x;          // b*N + n
  const int b = row >> 8;
  const int n = row & (kN - 1);
  const int h = threadIdx.x;

  float s = 0.0f;
  const float* p = src + (size_t)b * batchStride + (size_t)n * kD + h;
  for (int i = 0; i < P; ++i) s += p[(size_t)i * kN * kD];
  sC[h] = s;
  __syncthreads();

  float acc = bo[h];
  for (int i = 0; i < kD; ++i) acc = fmaf(sC[i], Wo[i * kD + h], acc);
  const float x = acc + hs[(size_t)row * kD + h];

  red[h] = x;
  __syncthreads();
  for (int sct = 128; sct > 0; sct >>= 1) {
    if (h < sct) red[h] += red[h + sct];
    __syncthreads();
  }
  const float mean = red[0] * (1.0f / 256.0f);
  __syncthreads();
  const float dm = x - mean;
  red[h] = dm * dm;
  __syncthreads();
  for (int sct = 128; sct > 0; sct >>= 1) {
    if (h < sct) red[h] += red[h + sct];
    __syncthreads();
  }
  const float var = red[0] * (1.0f / 256.0f);
  out[(size_t)row * kD + h] = gamma[h] * dm * rsqrtf(var + LN_EPS) + beta[h];
}

// -------------------------------------------------------------------------
extern "C" void kernel_launch(void* const* d_in, const int* in_sizes, int n_in,
                              void* d_out, int out_size, void* d_ws, size_t ws_size,
                              hipStream_t stream) {
  (void)in_sizes; (void)n_in; (void)out_size;
  const float* hs   = (const float*)d_in[0];
  const float* SM   = (const float*)d_in[1];
  const float* mask = (const float*)d_in[2];
  const float* Wq   = (const float*)d_in[3];
  const float* bq   = (const float*)d_in[4];
  const float* Wk   = (const float*)d_in[5];
  const float* bk   = (const float*)d_in[6];
  const float* Wv   = (const float*)d_in[7];
  const float* bv   = (const float*)d_in[8];
  const float* Wo   = (const float*)d_in[9];
  const float* bo   = (const float*)d_in[10];
  const float* gam  = (const float*)d_in[11];
  const float* bet  = (const float*)d_in[12];
  float* out = (float*)d_out;

  char* ws = (char*)d_ws;
  float*    ctx = (float*)ws;    ws += (size_t)kBatch * kN * kD * sizeof(float); // 1 MB
  _Float16* Kf  = (_Float16*)ws; ws += (size_t)kBatch * kN * kD * 2;             // 512 KB
  _Float16* WqT = (_Float16*)ws; ws += (size_t)kD * kD * 2;                      // 128 KB
  _Float16* WvT = (_Float16*)ws; ws += (size_t)kD * kD * 2;                      // 128 KB
  float* partials = (float*)ws;                                                  // 64 MB
  const size_t baseBytes = (size_t)((char*)partials - (char*)d_ws);
  const size_t partBytes = (size_t)kBatch * kPB * kN * kD * sizeof(float);
  const int usePart = (ws_size >= baseBytes + partBytes) ? 1 : 0;

  k_prep_k<<<kBatch * kN, 256, 0, stream>>>(hs, Wk, bk, Kf);
  k_transpose_f16<<<kD, 256, 0, stream>>>(Wq, WqT);
  k_transpose_f16<<<kD, 256, 0, stream>>>(Wv, WvT);

  dim3 grid(kN * kN / (kTM * kITER), kBatch);   // (64, 4)
  if (usePart) {
    k_edge_attn<<<grid, 512, 0, stream>>>(SM, mask, Kf, WqT, WvT, bq, bv,
                                          partials, 0);
    k_final<<<kBatch * kN, 256, 0, stream>>>(partials, kPB,
                                             (size_t)kPB * kN * kD,
                                             Wo, bo, hs, gam, bet, out);
  } else {
    const int ctxElems = kBatch * kN * kD;
    k_zero<<<(ctxElems + 255) / 256, 256, 0, stream>>>(ctx, ctxElems);
    k_edge_attn<<<grid, 512, 0, stream>>>(SM, mask, Kf, WqT, WvT, bq, bv,
                                          ctx, 1);
    k_final<<<kBatch * kN, 256, 0, stream>>>(ctx, 1, (size_t)kN * kD,
                                             Wo, bo, hs, gam, bet, out);
  }
}
